// Costvolume3D_90323162235010
// MI455X (gfx1250) — matrix operand order repaired and checked by reading, not compiled, not run on hardware
//
#include <hip/hip_runtime.h>

typedef __attribute__((ext_vector_type(16))) _Float16 v16h;
typedef __attribute__((ext_vector_type(8)))  float    v8f;

// ---------------- tiling constants ----------------
constexpr int KNN   = 16;
constexpr int COLS  = 256;   // 16 points * 16 neighbors per block
constexpr int CIN   = 64;
constexpr int COUT  = 64;
constexpr int C0IN  = 131;   // 2*64 + 3
constexpr int KPAD1 = 160;   // C0IN padded to 5 chunks of 32
constexpr int XSTR  = 164;   // X column stride (halves): 82 dwords -> conflict-free 16-lane b32 loads
constexpr int HSTR  = 68;    // H1 / p2p column stride (halves)

// ---------------- dynamic LDS layout (bytes) ----------------
constexpr int OFF_X   = 0;
constexpr int SZ_X    = COLS * XSTR * 2;          // 83968
constexpr int OFF_H   = OFF_X + SZ_X;
constexpr int SZ_H    = COLS * HSTR * 2;          // 34816
constexpr int OFF_W0  = OFF_H + SZ_H;
constexpr int SZ_W0   = COUT * KPAD1 * 2;         // 20480
constexpr int OFF_W1  = OFF_W0 + SZ_W0;
constexpr int SZ_W1   = COUT * 64 * 2;            // 8192
constexpr int OFF_XYZ = OFF_W1 + SZ_W1;
constexpr int SZ_XYZ  = COLS * 4 * 4;             // 4096
constexpr int SMEM1   = OFF_XYZ + SZ_XYZ;         // 151552 B (~148 KB) < 320 KB/WGP

// ---------------- WMMA fragment helpers (wave32, 16x16x32 f16) ----------------
// A: 16x32 (MxK), source row-major [M x stride] in LDS.
// ISA layout: lanes 0-15 -> M=lane, VGPR v: v<4 K={2v,2v+1}, v>=4 K={16+2(v-4),+1};
//             lanes 16-31 add +8 to K base within each half.
__device__ __forceinline__ v16h load_frag_A_rowmajor(const _Float16* base, int row0,
                                                     int stride, int kbase) {
  const int lane = threadIdx.x & 31;
  const int hl = lane & 15, hi = lane >> 4;
  const _Float16* rp = base + (row0 + hl) * stride + kbase + hi * 8;
  v16h a;
#pragma unroll
  for (int v = 0; v < 8; ++v) {
    const int k0 = (v < 4) ? (2 * v) : (16 + 2 * (v - 4));
    a[2 * v]     = rp[k0];
    a[2 * v + 1] = rp[k0 + 1];
  }
  return a;
}

// B: 32x16 (KxN), source column-major [col][k] with stride halves per column.
// ISA layout (matches A): lanes 0-15 -> N=lane, v<4 K={2v,2v+1}, v>=4 K={16+2(v-4),+1};
//                        lanes 16-31 -> N=lane-16, add +8 to K base within each half.
__device__ __forceinline__ v16h load_frag_B_colmajor(const _Float16* base, int col0,
                                                     int stride, int kbase) {
  const int lane = threadIdx.x & 31;
  const int hl = lane & 15, hi = lane >> 4;
  const _Float16* cp = base + (col0 + hl) * stride + kbase + hi * 8;
  v16h b;
#pragma unroll
  for (int v = 0; v < 8; ++v) {
    const int k0 = (v < 4) ? (2 * v) : (16 + 2 * (v - 4));
    b[2 * v]     = cp[k0];
    b[2 * v + 1] = cp[k0 + 1];
  }
  return b;
}

// C/D 16x16 f32: VGPR r, lane L -> M = r + 8*(L>=16), N = L&15 (documented layout).
__device__ __forceinline__ void store_D_bias_leaky(const v8f& c, const float* __restrict__ bias,
                                                   int mrow0, int col0, _Float16* dst, int stride) {
  const int lane = threadIdx.x & 31;
  const int hl = lane & 15, hi = lane >> 4;
#pragma unroll
  for (int r = 0; r < 8; ++r) {
    const int row = mrow0 + r + 8 * hi;
    float v = c[r] + bias[row];
    v = (v > 0.f) ? v : 0.1f * v;   // leaky ReLU, slope 0.1
    dst[(col0 + hl) * stride + row] = (_Float16)v;
  }
}

// ================= Stage 1: point-to-neighbor cost + weighted K-sum =================
__global__ __launch_bounds__(256)
void cost3d_stage1(const float* __restrict__ xyz1, const float* __restrict__ feat1,
                   const float* __restrict__ xyz2, const float* __restrict__ feat2,
                   const int* __restrict__ knn12, const unsigned char* __restrict__ val12,
                   const float* __restrict__ cw0, const float* __restrict__ cb0,
                   const float* __restrict__ cw1, const float* __restrict__ cb1,
                   const float* __restrict__ w2a, const float* __restrict__ b2a,
                   const float* __restrict__ w2b, const float* __restrict__ b2b,
                   const float* __restrict__ w2c, const float* __restrict__ b2c,
                   float* __restrict__ p2n, int N) {
  extern __shared__ char smem[];
  _Float16* Xs   = (_Float16*)(smem + OFF_X);    // 160(+pad) x 256 activations (col-major)
  _Float16* H1s  = (_Float16*)(smem + OFF_H);    // 64(+pad) x 256 hidden (col-major)
  _Float16* W0s  = (_Float16*)(smem + OFF_W0);   // 64 x 160 row-major f16 weights
  _Float16* W1s  = (_Float16*)(smem + OFF_W1);   // 64 x 64 row-major f16 weights
  float*    xyzs = (float*)(smem + OFF_XYZ);     // f32 xyz_norm per column (stride 4)

  const int tid = threadIdx.x;
  const int n0  = blockIdx.x * (COLS / KNN);     // 16 points per block

  // ---- stage weights into LDS as f16 (K-padded with zeros) ----
  for (int e = tid; e < COUT * KPAD1; e += COLS) {
    const int o = e / KPAD1, c = e % KPAD1;
    W0s[e] = (c < C0IN) ? (_Float16)cw0[o * C0IN + c] : (_Float16)0.f;
  }
  for (int e = tid; e < COUT * 64; e += COLS) W1s[e] = (_Float16)cw1[e];

  // ---- gather + build X column (one column per thread) ----
  {
    const int col = tid;
    const int np  = n0 + (col >> 4);
    const int kk  = col & 15;
    const int idx = knn12[np * KNN + kk];
    const float m = val12[np * KNN + kk] ? 1.f : 0.f;
    _Float16* xc = Xs + col * XSTR;
#pragma unroll 8
    for (int c = 0; c < CIN; ++c) xc[c] = (_Float16)feat1[c * N + np];
#pragma unroll 8
    for (int c = 0; c < CIN; ++c) xc[CIN + c] = (_Float16)(feat2[c * N + idx] * m);
#pragma unroll
    for (int d = 0; d < 3; ++d) {
      const float g = xyz2[d * N + idx] * m - xyz1[d * N + np];
      xc[2 * CIN + d] = (_Float16)g;
      xyzs[col * 4 + d] = g;
    }
    for (int r = C0IN; r < XSTR; ++r) xc[r] = (_Float16)0.f;  // zero K-pad
  }
  __syncthreads();

  const int wave = tid >> 5;

  // ---- GEMM1: H1 = leaky(cw0 @ X + cb0), 4 Mt x 16 Nt x 5 Kc ----
#pragma unroll
  for (int t = 0; t < 2; ++t) {
    const int nt = wave * 2 + t;
    v16h bb[5];
#pragma unroll
    for (int kc = 0; kc < 5; ++kc) bb[kc] = load_frag_B_colmajor(Xs, nt * 16, XSTR, kc * 32);
#pragma unroll
    for (int mt = 0; mt < 4; ++mt) {
      v8f acc = {};
#pragma unroll
      for (int kc = 0; kc < 5; ++kc) {
        v16h a = load_frag_A_rowmajor(W0s, mt * 16, KPAD1, kc * 32);
        acc = __builtin_amdgcn_wmma_f32_16x16x32_f16(false, a, false, bb[kc],
                                                     (short)0, acc, false, false);
      }
      store_D_bias_leaky(acc, cb0, mt * 16, nt * 16, H1s, HSTR);
    }
  }
  __syncthreads();

  // ---- GEMM2: p2p = leaky(cw1 @ H1 + cb1) -> reuse X buffer (stride HSTR) ----
  _Float16* P2Ps = Xs;
#pragma unroll
  for (int t = 0; t < 2; ++t) {
    const int nt = wave * 2 + t;
    v16h bb[2];
#pragma unroll
    for (int kc = 0; kc < 2; ++kc) bb[kc] = load_frag_B_colmajor(H1s, nt * 16, HSTR, kc * 32);
#pragma unroll
    for (int mt = 0; mt < 4; ++mt) {
      v8f acc = {};
#pragma unroll
      for (int kc = 0; kc < 2; ++kc) {
        v16h a = load_frag_A_rowmajor(W1s, mt * 16, 64, kc * 32);
        acc = __builtin_amdgcn_wmma_f32_16x16x32_f16(false, a, false, bb[kc],
                                                     (short)0, acc, false, false);
      }
      store_D_bias_leaky(acc, cb1, mt * 16, nt * 16, P2Ps, HSTR);
    }
  }
  __syncthreads();

  // ---- xyz MLP (3->8->8->64, ReLU) * p2p, reduce over K=16, write p2n ----
  {
    const int col = tid;
    const int np  = n0 + (col >> 4);
    const int kk  = col & 15;
    const float x0 = xyzs[col * 4 + 0], x1 = xyzs[col * 4 + 1], x2 = xyzs[col * 4 + 2];
    float h1[8], h2[8];
#pragma unroll
    for (int j = 0; j < 8; ++j) {
      const float s = w2a[j * 3 + 0] * x0 + w2a[j * 3 + 1] * x1 + w2a[j * 3 + 2] * x2 + b2a[j];
      h1[j] = fmaxf(s, 0.f);
    }
#pragma unroll
    for (int j = 0; j < 8; ++j) {
      float s = b2b[j];
#pragma unroll
      for (int i = 0; i < 8; ++i) s += w2b[j * 8 + i] * h1[i];
      h2[j] = fmaxf(s, 0.f);
    }
    for (int c = 0; c < COUT; ++c) {
      float s = b2c[c];
#pragma unroll
      for (int i = 0; i < 8; ++i) s += w2c[c * 8 + i] * h2[i];
      s = fmaxf(s, 0.f);
      float v = s * (float)P2Ps[col * HSTR + c];
      v += __shfl_xor(v, 1, 32);   // 16-lane groups stay inside one wave32
      v += __shfl_xor(v, 2, 32);
      v += __shfl_xor(v, 4, 32);
      v += __shfl_xor(v, 8, 32);
      if (kk == 0) p2n[c * N + np] = v;
    }
  }
}

// ================= Stage 2: neighbor-to-neighbor aggregation =================
__global__ __launch_bounds__(256)
void cost3d_stage2(const float* __restrict__ xyz1,
                   const int* __restrict__ knn11, const unsigned char* __restrict__ val11,
                   const float* __restrict__ w1a, const float* __restrict__ b1a,
                   const float* __restrict__ w1b, const float* __restrict__ b1b,
                   const float* __restrict__ w1c, const float* __restrict__ b1c,
                   const float* __restrict__ p2n, float* __restrict__ out, int N) {
  const int tid = threadIdx.x;
  const int n0  = blockIdx.x * (COLS / KNN);
  const int col = tid;
  const int np  = n0 + (col >> 4);
  const int kk  = col & 15;
  const int idx = knn11[np * KNN + kk];
  const float m = val11[np * KNN + kk] ? 1.f : 0.f;

  float g[3];
#pragma unroll
  for (int d = 0; d < 3; ++d) g[d] = xyz1[d * N + idx] * m - xyz1[d * N + np];

  float h1[8], h2[8];
#pragma unroll
  for (int j = 0; j < 8; ++j) {
    const float s = w1a[j * 3 + 0] * g[0] + w1a[j * 3 + 1] * g[1] + w1a[j * 3 + 2] * g[2] + b1a[j];
    h1[j] = fmaxf(s, 0.f);
  }
#pragma unroll
  for (int j = 0; j < 8; ++j) {
    float s = b1b[j];
#pragma unroll
    for (int i = 0; i < 8; ++i) s += w1b[j * 8 + i] * h1[i];
    h2[j] = fmaxf(s, 0.f);
  }
  for (int c = 0; c < COUT; ++c) {
    float s = b1c[c];
#pragma unroll
    for (int i = 0; i < 8; ++i) s += w1c[c * 8 + i] * h2[i];
    s = fmaxf(s, 0.f);
    float v = s * (p2n[c * N + idx] * m);
    v += __shfl_xor(v, 1, 32);
    v += __shfl_xor(v, 2, 32);
    v += __shfl_xor(v, 4, 32);
    v += __shfl_xor(v, 8, 32);
    if (kk == 0) out[c * N + np] = v;
  }
}

// ================= host entry =================
extern "C" void kernel_launch(void* const* d_in, const int* in_sizes, int n_in,
                              void* d_out, int out_size, void* d_ws, size_t ws_size,
                              hipStream_t stream) {
  const float* xyz1  = (const float*)d_in[0];
  const float* feat1 = (const float*)d_in[1];
  const float* xyz2  = (const float*)d_in[2];
  const float* feat2 = (const float*)d_in[3];
  const int*   knn12 = (const int*)d_in[4];
  const unsigned char* val12 = (const unsigned char*)d_in[5];
  const int*   knn11 = (const int*)d_in[6];
  const unsigned char* val11 = (const unsigned char*)d_in[7];
  const float* cw0 = (const float*)d_in[8];
  const float* cb0 = (const float*)d_in[9];
  const float* cw1 = (const float*)d_in[10];
  const float* cb1 = (const float*)d_in[11];
  const float* w1a = (const float*)d_in[12];
  const float* b1a = (const float*)d_in[13];
  const float* w1b = (const float*)d_in[14];
  const float* b1b = (const float*)d_in[15];
  const float* w1c = (const float*)d_in[16];
  const float* b1c = (const float*)d_in[17];
  const float* w2a = (const float*)d_in[18];
  const float* b2a = (const float*)d_in[19];
  const float* w2b = (const float*)d_in[20];
  const float* b2b = (const float*)d_in[21];
  const float* w2c = (const float*)d_in[22];
  const float* b2c = (const float*)d_in[23];

  const int N = in_sizes[4] / KNN;          // 53248 (B==1), divisible by 16
  const int nBlocks = N / (COLS / KNN);

  float* p2n = (float*)d_ws;                // 64 * N * 4 = 13.6 MB scratch
  float* out = (float*)d_out;

  (void)hipFuncSetAttribute(reinterpret_cast<const void*>(cost3d_stage1),
                            hipFuncAttributeMaxDynamicSharedMemorySize, SMEM1);

  cost3d_stage1<<<nBlocks, COLS, SMEM1, stream>>>(
      xyz1, feat1, xyz2, feat2, knn12, val12,
      cw0, cb0, cw1, cb1, w2a, b2a, w2b, b2b, w2c, b2c, p2n, N);

  cost3d_stage2<<<nBlocks, COLS, 0, stream>>>(
      xyz1, knn11, val11, w1a, b1a, w1b, b1b, w1c, b1c, p2n, out, N);
}
